// GNNModel_80023830659516
// MI455X (gfx1250) — matrix operand order repaired and checked by reading, not compiled, not run on hardware
//
#include <hip/hip_runtime.h>
#include <hip/hip_bf16.h>

// ---------- types ----------
typedef __bf16 bf16;
typedef __attribute__((ext_vector_type(16))) __bf16 v16bf;
typedef __attribute__((ext_vector_type(8)))  float  v8f;
typedef unsigned int u32x4 __attribute__((ext_vector_type(4)));
typedef float f32x4 __attribute__((ext_vector_type(4)));
typedef float f32x2 __attribute__((ext_vector_type(2)));

union FragBf { v16bf v; u32x4 q[2]; };

// ---------- small elementwise kernels ----------
__global__ void k_set_f32(float* __restrict__ p, float val, int n) {
  int i = blockIdx.x * blockDim.x + threadIdx.x;
  if (i < n) p[i] = val;
}

__global__ void k_degree(const long long* __restrict__ dst, float* __restrict__ deg, int E) {
  int i = blockIdx.x * blockDim.x + threadIdx.x;
  if (i < E) unsafeAtomicAdd(&deg[(int)dst[i]], 1.0f);
}

__global__ void k_rsqrt(const float* __restrict__ deg, float* __restrict__ dinv, int n) {
  int i = blockIdx.x * blockDim.x + threadIdx.x;
  if (i < n) {
    float d = deg[i];
    dinv[i] = d > 0.f ? rsqrtf(d) : 0.f;
  }
}

__global__ void k_f32_to_bf16(const float* __restrict__ in, bf16* __restrict__ out, size_t n) {
  size_t i = (size_t)blockIdx.x * blockDim.x + threadIdx.x;
  if (i < n) out[i] = (bf16)in[i];
}

// W[K x N] f32 (row-major) -> Wt[N x K] bf16 (row-major): B-fragment loads become contiguous.
__global__ void k_transpose_cvt(const float* __restrict__ W, bf16* __restrict__ Wt, int K, int Ncol) {
  int i = blockIdx.x * blockDim.x + threadIdx.x;
  if (i < K * Ncol) {
    int k = i / Ncol, nc = i % Ncol;
    Wt[(size_t)nc * K + k] = (bf16)W[i];
  }
}

__global__ void k_bias_relu_cvt(const float* __restrict__ a, const float* __restrict__ b,
                                bf16* __restrict__ out, size_t n, int mask) {
  size_t i = (size_t)blockIdx.x * blockDim.x + threadIdx.x;
  if (i < n) {
    float v = a[i] + b[i & (size_t)mask];
    out[i] = (bf16)fmaxf(v, 0.f);
  }
}

__global__ void k_bias_out(const float* __restrict__ a, const float* __restrict__ b,
                           float* __restrict__ out, size_t n, int mask) {
  size_t i = (size_t)blockIdx.x * blockDim.x + threadIdx.x;
  if (i < n) out[i] = a[i] + b[i & (size_t)mask];
}

// ---------- WMMA GEMM: C[nrows x NDIM] = A[nrows x 128](bf16) * Bt[NDIM x 128](bf16)^T ----------
// One wave per 16x16 output tile; K=128 unrolled into 4 x v_wmma_f32_16x16x32_bf16.
template <int NDIM>
__global__ void k_gemm_bf16(const bf16* __restrict__ A, const bf16* __restrict__ Bt,
                            float* __restrict__ C, int nrows) {
  constexpr int K  = 128;
  constexpr int NT = NDIM / 16;
  int wave = (int)((blockIdx.x * blockDim.x + threadIdx.x) >> 5);
  int lane = threadIdx.x & 31;
  int mtiles = (nrows + 15) >> 4;
  if (wave >= mtiles * NT) return;                 // wave-uniform: EXEC stays all-ones
  int mtile = wave / NT, ntile = wave % NT;
  int half = lane >> 4, l15 = lane & 15;

  int mrow = mtile * 16 + l15;
  int mclamp = mrow < nrows ? mrow : nrows - 1;    // safe A reads for tail tiles
  const bf16* arow = A  + (size_t)mclamp * K;
  const bf16* brow = Bt + (size_t)(ntile * 16 + l15) * K;

  v8f acc = {};
#pragma unroll
  for (int kb = 0; kb < K; kb += 32) {
    FragBf fa, fb;
    // A 16x32 bf16 fragment (ISA 7.12.2): lane half selects K sub-block of 8.
    fa.q[0] = *(const u32x4*)(arow + kb + 8 * half);        // K = kb + 8*half + [0..7]
    fa.q[1] = *(const u32x4*)(arow + kb + 16 + 8 * half);   // K = kb + 16 + 8*half + [0..7]
    // B 32x16 bf16 fragment: lane = column, same K sub-blocks of 8 as A.
    const u32x4* bp = (const u32x4*)(brow + kb + 8 * half);
    fb.q[0] = bp[0];
    fb.q[1] = bp[2];
    acc = __builtin_amdgcn_wmma_f32_16x16x32_bf16(false, fa.v, false, fb.v,
                                                  (short)0, acc, false, false);
  }
  // D layout: col = lane&15, row = vgpr + 8*(lane>=16)
  float* cp = C + (size_t)(mtile * 16 + 8 * half) * NDIM + ntile * 16 + l15;
  if (mtile * 16 + 16 <= nrows) {
    // full tile (the common case; nrows % 16 == 0 here): unguarded stores
#pragma unroll
    for (int v = 0; v < 8; ++v) cp[(size_t)v * NDIM] = acc[v];
  } else {
#pragma unroll
    for (int v = 0; v < 8; ++v) {
      int m = mtile * 16 + v + 8 * half;
      if (m < nrows) cp[(size_t)v * NDIM] = acc[v];
    }
  }
}

// ---------- edge aggregation: out[dst] += h[src] * dinv[src]*dinv[dst] (+self loops) ----------
// One wave per edge; lane handles VPL consecutive floats -> one b128/b64 row load per lane,
// native global_atomic_add_f32 scatters.
template <int F, int VPL>
__global__ void k_aggregate(const float* __restrict__ h,
                            const long long* __restrict__ src,
                            const long long* __restrict__ dst,
                            const float* __restrict__ dinv,
                            float* __restrict__ out, int E, int n) {
  int wave = (int)((blockIdx.x * blockDim.x + threadIdx.x) >> 5);
  int lane = threadIdx.x & 31;
  int total = E + n;
  if (wave >= total) return;
  int s, d;
  if (wave < E) { s = (int)src[wave]; d = (int)dst[wave]; }
  else          { s = d = wave - E; }               // self loop
  float norm = dinv[s] * dinv[d];
  const float* hp = h + (size_t)s * F + lane * VPL;
  float* op = out + (size_t)d * F + lane * VPL;
  if constexpr (VPL == 4) {
    f32x4 t = *(const f32x4*)hp;                    // global_load_b128
#pragma unroll
    for (int i = 0; i < 4; ++i) unsafeAtomicAdd(op + i, t[i] * norm);
  } else {
    f32x2 t = *(const f32x2*)hp;                    // global_load_b64
#pragma unroll
    for (int i = 0; i < 2; ++i) unsafeAtomicAdd(op + i, t[i] * norm);
  }
}

// ---------- host orchestration ----------
extern "C" void kernel_launch(void* const* d_in, const int* in_sizes, int n_in,
                              void* d_out, int out_size, void* d_ws, size_t ws_size,
                              hipStream_t stream) {
  const float*     x  = (const float*)d_in[0];
  const long long* ei = (const long long*)d_in[1];   // int64 [2 x E]
  const float*     W1 = (const float*)d_in[2];
  const float*     b1 = (const float*)d_in[3];
  const float*     W2 = (const float*)d_in[4];
  const float*     b2 = (const float*)d_in[5];
  float* out = (float*)d_out;

  const int IN = 128, HID = 128, OUT = 64;
  const int nN = in_sizes[0] / IN;          // 100000
  const int E  = in_sizes[1] / 2;           // 1600000
  const long long* srcA = ei;
  const long long* dstA = ei + E;

  auto al = [](size_t v) { return (v + 255) & ~(size_t)255; };
  char* ws = (char*)d_ws;
  size_t o = 0;
  float* deg  = (float*)(ws + o); o = al(o + (size_t)nN * 4);
  float* dinv = (float*)(ws + o); o = al(o + (size_t)nN * 4);
  bf16*  W1t  = (bf16*)(ws + o);  o = al(o + (size_t)IN * HID * 2);
  bf16*  W2t  = (bf16*)(ws + o);  o = al(o + (size_t)HID * OUT * 2);
  char*  xbfR = ws + o;           o = al(o + (size_t)nN * IN * 2);   // x_bf; reused as agg2 (nN*OUT*4, same bytes)
  char*  bufA = ws + o;           o = al(o + (size_t)nN * HID * 4);  // h1 f32; reused as h2_bf
  char*  bufB = ws + o;           o = al(o + (size_t)nN * HID * 4);  // agg1 f32; reused as g2 f32
  (void)ws_size;

  bf16*  x_bf  = (bf16*)xbfR;
  float* h1    = (float*)bufA;
  float* agg1  = (float*)bufB;
  bf16*  h2_bf = (bf16*)bufA;
  float* g2    = (float*)bufB;
  float* agg2  = (float*)xbfR;

  const int B = 256;
  auto blocks = [&](size_t work) { return (int)((work + B - 1) / B); };

  // 1) degrees (init 1.0 = self-loop) and d^-1/2
  k_set_f32<<<blocks(nN), B, 0, stream>>>(deg, 1.0f, nN);
  k_degree<<<blocks(E), B, 0, stream>>>(dstA, deg, E);
  k_rsqrt<<<blocks(nN), B, 0, stream>>>(deg, dinv, nN);

  // 2) bf16 conversions
  k_f32_to_bf16<<<blocks((size_t)nN * IN), B, 0, stream>>>(x, x_bf, (size_t)nN * IN);
  k_transpose_cvt<<<blocks((size_t)IN * HID), B, 0, stream>>>(W1, W1t, IN, HID);
  k_transpose_cvt<<<blocks((size_t)HID * OUT), B, 0, stream>>>(W2, W2t, HID, OUT);

  // 3) layer 1: GEMM (WMMA), scatter-aggregate, bias+ReLU+cvt
  {
    size_t waves = (size_t)((nN + 15) / 16) * (HID / 16);
    k_gemm_bf16<128><<<blocks(waves * 32), B, 0, stream>>>(x_bf, W1t, h1, nN);
  }
  hipMemsetAsync(agg1, 0, (size_t)nN * HID * 4, stream);
  {
    size_t waves = (size_t)E + nN;
    k_aggregate<128, 4><<<blocks(waves * 32), B, 0, stream>>>(h1, srcA, dstA, dinv, agg1, E, nN);
  }
  k_bias_relu_cvt<<<blocks((size_t)nN * HID), B, 0, stream>>>(agg1, b1, h2_bf, (size_t)nN * HID, HID - 1);

  // 4) layer 2: GEMM (WMMA), scatter-aggregate, bias
  {
    size_t waves = (size_t)((nN + 15) / 16) * (OUT / 16);
    k_gemm_bf16<64><<<blocks(waves * 32), B, 0, stream>>>(h2_bf, W2t, g2, nN);
  }
  hipMemsetAsync(agg2, 0, (size_t)nN * OUT * 4, stream);   // x_bf dead after GEMM1
  {
    size_t waves = (size_t)E + nN;
    k_aggregate<64, 2><<<blocks(waves * 32), B, 0, stream>>>(g2, srcA, dstA, dinv, agg2, E, nN);
  }
  k_bias_out<<<blocks((size_t)nN * OUT), B, 0, stream>>>(agg2, b2, out, (size_t)nN * OUT, OUT - 1);
}
